// EnhancedQuerySelector_8349416423987
// MI455X (gfx1250) — matrix operand order repaired and checked by reading, not compiled, not run on hardware
//
#include <hip/hip_runtime.h>
#include <hip/hip_bf16.h>
#include <math.h>

#define B_   8
#define NI_  4096
#define NT_  512
#define H_   768
#define D_   64
#define Q_   100
#define LN_EPS 1e-5f

typedef __attribute__((ext_vector_type(16))) __bf16 bf16x16;
typedef __attribute__((ext_vector_type(8)))  float  f32x8;

__device__ __forceinline__ float gelu_f(float x) {
    // exact gelu (approximate=False): 0.5*x*(1+erf(x/sqrt(2)))
    return 0.5f * x * (1.0f + erff(x * 0.7071067811865476f));
}
__device__ __forceinline__ float wred_sum(float v) {
    #pragma unroll
    for (int o = 16; o > 0; o >>= 1) v += __shfl_xor(v, o, 32);
    return v;
}
__device__ __forceinline__ float wred_max(float v) {
    #pragma unroll
    for (int o = 16; o > 0; o >>= 1) v = fmaxf(v, __shfl_xor(v, o, 32));
    return v;
}

// ---------------------------------------------------------------------------
// LayerNorm over H=768 per row -> bf16 (optional raw bf16 copy for tw path)
// ---------------------------------------------------------------------------
__global__ void k_layernorm(const float* __restrict__ x, const float* __restrict__ g,
                            const float* __restrict__ be, __bf16* __restrict__ lnout,
                            __bf16* __restrict__ rawout) {
    int row = blockIdx.x;
    const float* xr = x + (size_t)row * H_;
    __shared__ float s1[256], s2[256];
    float a = 0.f, b = 0.f;
    for (int h = threadIdx.x; h < H_; h += 256) { float v = xr[h]; a += v; b += v * v; }
    s1[threadIdx.x] = a; s2[threadIdx.x] = b; __syncthreads();
    for (int st = 128; st > 0; st >>= 1) {
        if (threadIdx.x < st) {
            s1[threadIdx.x] += s1[threadIdx.x + st];
            s2[threadIdx.x] += s2[threadIdx.x + st];
        }
        __syncthreads();
    }
    float m   = s1[0] * (1.0f / H_);
    float var = s2[0] * (1.0f / H_) - m * m;          // biased var (jnp.var)
    float inv = rsqrtf(var + LN_EPS);
    for (int h = threadIdx.x; h < H_; h += 256) {
        float v = xr[h];
        lnout[(size_t)row * H_ + h] = (__bf16)((v - m) * inv * g[h] + be[h]);
        if (rawout) rawout[(size_t)row * H_ + h] = (__bf16)v;
    }
}

// ---------------------------------------------------------------------------
// Pack 768x64 f32 weights into bf16 fragment-major order:
// Wp[kt][j][lane][e] = W[kt*32 + (lane>>4)*16 + e][j*16 + (lane&15)]
// -> each lane's 16 bf16 B-fragment elements are 32 contiguous bytes.
// ---------------------------------------------------------------------------
__global__ void k_pack_w(const float* __restrict__ W, __bf16* __restrict__ Wp) {
    int idx = blockIdx.x * 256 + threadIdx.x;          // 24*4*32*16 = 49152
    if (idx >= (H_ / 32) * 4 * 32 * 16) return;
    int e    = idx & 15;
    int lane = (idx >> 4) & 31;
    int j    = (idx >> 9) & 3;
    int kt   = idx >> 11;
    int l = lane & 15, hi = lane >> 4;
    Wp[idx] = (__bf16)W[(size_t)(kt * 32 + hi * 16 + e) * D_ + j * 16 + l];
}

// ---------------------------------------------------------------------------
// WMMA GEMM: [Mrows,768](bf16) x packed[768,64](bf16) -> epilogue.
// Wave computes 16x64 (4 C tiles, shared A fragment). Ping-pong double buffer
// (K unrolled by 2) -> no inter-buffer register copies.
// block = 256 (8 waves) -> 128 rows/block. grid = Mrows/128.
//   mode 0: gelu(acc+bias)+resid -> f32   mode 1: gelu(acc+bias) -> f32
//   mode 2: gelu(acc+bias)+resid -> bf16
// ---------------------------------------------------------------------------
__global__ void k_gemm768x64(const __bf16* __restrict__ A, const __bf16* __restrict__ Wp,
                             const float* __restrict__ bias, const float* __restrict__ resid,
                             float* __restrict__ outf, __bf16* __restrict__ outb, int mode) {
    int wave = threadIdx.x >> 5;
    int lane = threadIdx.x & 31;
    int l = lane & 15, hi = lane >> 4;
    int m0 = blockIdx.x * 128 + wave * 16;

    const __bf16* arow = A + (size_t)(m0 + l) * H_;
    f32x8 acc[4] = {};

    auto loadA = [&](int kk, bf16x16& a) {
        #pragma unroll
        for (int t = 0; t < 8; t++) a[t]     = arow[kk + hi * 16 + t];
        #pragma unroll
        for (int t = 0; t < 8; t++) a[8 + t] = arow[kk + hi * 16 + 8 + t];
    };
    auto loadB = [&](int kt, bf16x16* bb) {
        #pragma unroll
        for (int j = 0; j < 4; j++) {
            const __bf16* src = Wp + ((size_t)(kt * 4 + j) * 32 + lane) * 16;
            #pragma unroll
            for (int e = 0; e < 16; e++) bb[j][e] = src[e];
        }
    };

    bf16x16 a0, a1, b0[4], b1[4];
    loadA(0, a0);
    loadB(0, b0);
    for (int kt = 0; kt < H_ / 32; kt += 2) {          // 24 steps, even
        // prefetch odd step while computing even step
        loadA((kt + 1) * 32, a1);
        loadB(kt + 1, b1);
        #pragma unroll
        for (int j = 0; j < 4; j++)
            acc[j] = __builtin_amdgcn_wmma_f32_16x16x32_bf16(false, a0, false, b0[j],
                                                             (short)0, acc[j], false, false);
        // prefetch next even step while computing odd step
        if (kt + 2 < H_ / 32) {
            loadA((kt + 2) * 32, a0);
            loadB(kt + 2, b0);
        }
        #pragma unroll
        for (int j = 0; j < 4; j++)
            acc[j] = __builtin_amdgcn_wmma_f32_16x16x32_bf16(false, a1, false, b1[j],
                                                             (short)0, acc[j], false, false);
    }
    #pragma unroll
    for (int j = 0; j < 4; j++) {
        #pragma unroll
        for (int r = 0; r < 8; r++) {
            int m = m0 + r + hi * 8;
            int n = j * 16 + l;
            float v = gelu_f(acc[j][r] + bias[n]);
            if (mode != 1) v += resid[(size_t)m * H_ + n];   // features[..., :64]
            if (mode == 2) outb[(size_t)m * D_ + n] = (__bf16)v;
            else           outf[(size_t)m * D_ + n] = v;
        }
    }
}

// tw tail: sigmoid(h @ tw_w2 + b2) per token
__global__ void k_tw2(const float* __restrict__ h, const float* __restrict__ w2,
                      const float* __restrict__ b2, float* __restrict__ tw_raw) {
    int row = blockIdx.x * blockDim.x + threadIdx.x;
    if (row >= B_ * NT_) return;
    float acc = 0.f;
    #pragma unroll 8
    for (int c = 0; c < D_; c++) acc = fmaf(h[(size_t)row * D_ + c], w2[c], acc);
    acc += b2[0];
    tw_raw[row] = 1.0f / (1.0f + expf(-acc));
}

// softmax of tw over NT per batch; weighted_text in f32 (for Gram) + bf16 (WMMA B)
__global__ void k_tw_softmax(const float* __restrict__ tw_raw,
                             const float* __restrict__ text_c,
                             float* __restrict__ wtf, __bf16* __restrict__ wtb) {
    int b = blockIdx.x;
    __shared__ float red[256];
    __shared__ float smax, ssum;
    const float* r = tw_raw + b * NT_;
    float m = -INFINITY;
    for (int t = threadIdx.x; t < NT_; t += 256) m = fmaxf(m, r[t]);
    red[threadIdx.x] = m; __syncthreads();
    for (int st = 128; st > 0; st >>= 1) {
        if (threadIdx.x < st) red[threadIdx.x] = fmaxf(red[threadIdx.x], red[threadIdx.x + st]);
        __syncthreads();
    }
    if (threadIdx.x == 0) smax = red[0];
    __syncthreads();
    float s = 0.f;
    for (int t = threadIdx.x; t < NT_; t += 256) s += expf(r[t] - smax);
    red[threadIdx.x] = s; __syncthreads();
    for (int st = 128; st > 0; st >>= 1) {
        if (threadIdx.x < st) red[threadIdx.x] += red[threadIdx.x + st];
        __syncthreads();
    }
    if (threadIdx.x == 0) ssum = red[0];
    __syncthreads();
    for (int e = threadIdx.x; e < NT_ * D_; e += 256) {
        int t = e / D_;
        float w = expf(r[t] - smax) / ssum;
        float v = text_c[(size_t)b * NT_ * D_ + e] * w;
        wtf[(size_t)b * NT_ * D_ + e] = v;
        wtb[(size_t)b * NT_ * D_ + e] = (__bf16)v;
    }
}

// ---------------------------------------------------------------------------
// logits[b] = img_c[b](4096x64) @ wt[b]^T(64x512), scaled by 1/(|T|+1e-6)
// wave computes 16x64 (4 C tiles). 2048 blocks x 256.
// ---------------------------------------------------------------------------
__global__ void k_logits(const __bf16* __restrict__ imgc, const __bf16* __restrict__ wt,
                         const float* __restrict__ temp, float* __restrict__ logits) {
    int gw   = blockIdx.x * 8 + (threadIdx.x >> 5);
    int lane = threadIdx.x & 31, l = lane & 15, hi = lane >> 4;
    int b  = gw >> 11;
    int r  = gw & 2047;
    int m0  = (r >> 3) * 16;
    int n0b = (r & 7) * 64;
    const __bf16* Abase = imgc + (size_t)b * NI_ * D_;
    const __bf16* Bbase = wt   + (size_t)b * NT_ * D_;

    f32x8 acc[4] = {};
    const __bf16* arow = Abase + (size_t)(m0 + l) * D_;
    #pragma unroll
    for (int kk = 0; kk < D_; kk += 32) {
        bf16x16 a;
        #pragma unroll
        for (int t = 0; t < 8; t++) a[t]     = arow[kk + hi * 16 + t];
        #pragma unroll
        for (int t = 0; t < 8; t++) a[8 + t] = arow[kk + hi * 16 + 8 + t];
        #pragma unroll
        for (int j = 0; j < 4; j++) {
            bf16x16 bm;
            const __bf16* brow = Bbase + (size_t)(n0b + j * 16 + l) * D_ + kk + hi * 16;
            #pragma unroll
            for (int e = 0; e < 16; e++) bm[e] = brow[e];
            acc[j] = __builtin_amdgcn_wmma_f32_16x16x32_bf16(false, a, false, bm,
                                                             (short)0, acc[j], false, false);
        }
    }
    float invT = 1.0f / (fabsf(temp[0]) + 1e-6f);
    float* out = logits + (size_t)b * NI_ * NT_;
    #pragma unroll
    for (int r8 = 0; r8 < 8; r8++) {
        int m = m0 + r8 + hi * 8;
        #pragma unroll
        for (int j = 0; j < 4; j++)
            out[(size_t)m * NT_ + n0b + j * 16 + l] = acc[j][r8] * invT;
    }
}

// per-(b,j) column: streaming max + sum(exp) over NI (softmax axis=1 stats)
__global__ void k_colstats(const float* __restrict__ logits,
                           float* __restrict__ colmax, float* __restrict__ colsum) {
    int b = blockIdx.x >> 9;
    int j = blockIdx.x & (NT_ - 1);
    const float* base = logits + (size_t)b * NI_ * NT_ + j;
    float m = -INFINITY, s = 0.f;
    for (int i = threadIdx.x; i < NI_; i += 256) {
        float v = base[(size_t)i * NT_];
        if (v > m) { s = s * expf(m - v) + 1.0f; m = v; }
        else       { s += expf(v - m); }
    }
    __shared__ float sm[256], ss[256];
    sm[threadIdx.x] = m; ss[threadIdx.x] = s; __syncthreads();
    for (int st = 128; st > 0; st >>= 1) {
        if (threadIdx.x < st) {
            float m2 = sm[threadIdx.x + st], s2 = ss[threadIdx.x + st];
            float M  = fmaxf(sm[threadIdx.x], m2);
            ss[threadIdx.x] = ss[threadIdx.x] * expf(sm[threadIdx.x] - M) + s2 * expf(m2 - M);
            sm[threadIdx.x] = M;
        }
        __syncthreads();
    }
    if (threadIdx.x == 0) { colmax[b * NT_ + j] = sm[0]; colsum[b * NT_ + j] = ss[0]; }
}

// per-row: scores = max_j softmax_col(l), x2 = ||row||^2, log_s; init sum_d/mask
__global__ void k_scores(const float* __restrict__ logits, const float* __restrict__ colmax,
                         const float* __restrict__ colsum, float* __restrict__ x2,
                         float* __restrict__ logs, float* __restrict__ scores,
                         float* __restrict__ sumd, int* __restrict__ mask) {
    int gw   = blockIdx.x * 8 + (threadIdx.x >> 5);
    int lane = threadIdx.x & 31;
    int b = gw >> 12;
    int i = gw & (NI_ - 1);
    const float* lr = logits + ((size_t)b * NI_ + i) * NT_;
    const float* cm = colmax + b * NT_;
    const float* cs = colsum + b * NT_;
    float mx = -INFINITY, sq = 0.f;
    #pragma unroll 4
    for (int t = 0; t < NT_ / 32; t++) {
        int j = lane + 32 * t;
        float v = lr[j];
        sq = fmaf(v, v, sq);
        mx = fmaxf(mx, expf(v - cm[j]) / cs[j]);
    }
    mx = wred_max(mx);
    sq = wred_sum(sq);
    if (lane == 0) {
        int idx = b * NI_ + i;
        x2[idx] = sq; scores[idx] = mx; logs[idx] = logf(mx);
        sumd[idx] = 0.f; mask[idx] = 0;
    }
}

// ---------------------------------------------------------------------------
// Gram matrix: Gb[b] = invT^2 * (wt^T wt)  [64x64] bf16. LDS-tiled over NT.
// ---------------------------------------------------------------------------
__global__ void __launch_bounds__(256)
k_gram(const float* __restrict__ wtf, const float* __restrict__ temp,
       __bf16* __restrict__ Gb) {
    int b = blockIdx.x;
    __shared__ float sw[64 * D_];                  // 16KB tile (64 tokens)
    float accv[16];
    #pragma unroll
    for (int q = 0; q < 16; q++) accv[q] = 0.f;
    for (int t0 = 0; t0 < NT_; t0 += 64) {
        for (int e = threadIdx.x; e < 64 * D_; e += 256)
            sw[e] = wtf[(size_t)b * NT_ * D_ + (size_t)t0 * D_ + e];
        __syncthreads();
        for (int tt = 0; tt < 64; tt++) {
            #pragma unroll
            for (int q = 0; q < 16; q++) {
                int e = threadIdx.x + 256 * q;
                accv[q] = fmaf(sw[tt * D_ + (e >> 6)], sw[tt * D_ + (e & 63)], accv[q]);
            }
        }
        __syncthreads();
    }
    float invT = 1.0f / (fabsf(temp[0]) + 1e-6f);
    float s2 = invT * invT;
    #pragma unroll
    for (int q = 0; q < 16; q++) {
        int e = threadIdx.x + 256 * q;
        Gb[(size_t)b * D_ * D_ + e] = (__bf16)(accv[q] * s2);
    }
}

// ---------------------------------------------------------------------------
// P[b] = img_c[b](4096x64 bf16) @ Gb[b](64x64 bf16) -> f32. WMMA, K=64.
// So <logits_i, logits_c> == P_i . img_c_c  (Gram factorization).
// ---------------------------------------------------------------------------
__global__ void k_gemmP(const __bf16* __restrict__ imgc, const __bf16* __restrict__ Gb,
                        float* __restrict__ P) {
    int gw   = blockIdx.x * 8 + (threadIdx.x >> 5);   // 2048 waves total
    int lane = threadIdx.x & 31, l = lane & 15, hi = lane >> 4;
    int b  = gw >> 8;                                  // 256 strips per batch
    int m0 = (gw & 255) * 16;
    const __bf16* arow = imgc + (size_t)b * NI_ * D_ + (size_t)(m0 + l) * D_;
    const __bf16* G    = Gb + (size_t)b * D_ * D_;
    f32x8 acc[4] = {};
    #pragma unroll
    for (int kk = 0; kk < D_; kk += 32) {
        bf16x16 a;
        #pragma unroll
        for (int t = 0; t < 8; t++) a[t]     = arow[kk + hi * 16 + t];
        #pragma unroll
        for (int t = 0; t < 8; t++) a[8 + t] = arow[kk + hi * 16 + 8 + t];
        #pragma unroll
        for (int j = 0; j < 4; j++) {
            bf16x16 bm;
            #pragma unroll
            for (int e = 0; e < 16; e++)
                bm[e] = G[(size_t)(kk + hi * 16 + e) * D_ + j * 16 + l];
            acc[j] = __builtin_amdgcn_wmma_f32_16x16x32_bf16(false, a, false, bm,
                                                             (short)0, acc[j], false, false);
        }
    }
    float* out = P + (size_t)b * NI_ * D_;
    #pragma unroll
    for (int r8 = 0; r8 < 8; r8++) {
        int m = m0 + r8 + hi * 8;
        #pragma unroll
        for (int j = 0; j < 4; j++)
            out[(size_t)m * D_ + j * 16 + l] = acc[j][r8];
    }
}

// first pick: Gumbel-max categorical over scores/sum (deterministic hash noise)
__global__ void k_first(const float* __restrict__ scores, float* __restrict__ outF,
                        int* __restrict__ cur, int* __restrict__ mask) {
    int b = blockIdx.x;
    __shared__ float red[256]; __shared__ int redi[256]; __shared__ float ssum;
    const float* sc = scores + b * NI_;
    float s = 0.f;
    for (int i = threadIdx.x; i < NI_; i += 256) s += sc[i];
    red[threadIdx.x] = s; __syncthreads();
    for (int st = 128; st > 0; st >>= 1) {
        if (threadIdx.x < st) red[threadIdx.x] += red[threadIdx.x + st];
        __syncthreads();
    }
    if (threadIdx.x == 0) ssum = red[0];
    __syncthreads();
    float invden = 1.0f / (ssum + 1e-6f);
    float bv = -INFINITY; int bi = -1;
    for (int i = threadIdx.x; i < NI_; i += 256) {
        unsigned int h = (unsigned int)(b * NI_ + i) * 0x9E3779B9u + 0x85ebca6bu;
        h ^= h >> 16; h *= 0x7feb352du; h ^= h >> 15; h *= 0x846ca68bu; h ^= h >> 16;
        float u   = (float)((h >> 8) + 1u) * (1.0f / 16777216.0f);
        float gum = -logf(-logf(u));
        float val = logf(sc[i] * invden + 1e-9f) + gum;
        if (val > bv) { bv = val; bi = i; }
    }
    red[threadIdx.x] = bv; redi[threadIdx.x] = bi; __syncthreads();
    if (threadIdx.x == 0) {
        float Bv = -INFINITY; int Bi = -1;
        for (int t = 0; t < 256; t++) if (red[t] > Bv) { Bv = red[t]; Bi = redi[t]; }
        outF[b * Q_ + 0] = (float)Bi;
        cur[b] = Bi;
        mask[b * NI_ + Bi] = 1;
    }
}

// ---------------------------------------------------------------------------
// One selection step via Gram factorization: dot_i = P_i . img_c[cur] (D=64),
// sum_d += sqrt(max(x2_i + x2_c - 2 dot, 0)); argmax(log_s + lam*sum_d/k).
// One block per batch; reads only P (8MB total) instead of logits (67MB).
// ---------------------------------------------------------------------------
__global__ void __launch_bounds__(1024)
k_step(const float* __restrict__ P, const __bf16* __restrict__ imgc,
       const float* __restrict__ x2, const float* __restrict__ logs,
       float* __restrict__ sumd, int* __restrict__ mask, int* __restrict__ cur,
       const float* __restrict__ lamp, float* __restrict__ outF, int k) {
    int b = blockIdx.x;
    __shared__ float srow[D_];
    __shared__ float rv[1024]; __shared__ int ri[1024];
    __shared__ int scur; __shared__ float sx2c;
    if (threadIdx.x == 0) { scur = cur[b]; sx2c = x2[b * NI_ + scur]; }
    __syncthreads();
    if (threadIdx.x < D_)
        srow[threadIdx.x] = (float)imgc[((size_t)b * NI_ + scur) * D_ + threadIdx.x];
    __syncthreads();

    float lam = lamp[0], x2c = sx2c, invk = 1.0f / (float)k;
    float bv = -INFINITY; int bi = -1;
    #pragma unroll
    for (int rr = 0; rr < NI_ / 1024; rr++) {
        int i = threadIdx.x + rr * 1024;
        const float* pr = P + ((size_t)b * NI_ + i) * D_;
        float p = 0.f;
        #pragma unroll
        for (int c = 0; c < D_; c++) p = fmaf(pr[c], srow[c], p);
        int idx = b * NI_ + i;
        float d  = sqrtf(fmaxf(x2[idx] + x2c - 2.0f * p, 0.0f));
        float sd = sumd[idx] + d;
        sumd[idx] = sd;
        float val = mask[idx] ? -INFINITY : (logs[idx] + lam * sd * invk);
        if (val > bv) { bv = val; bi = i; }
    }
    rv[threadIdx.x] = bv; ri[threadIdx.x] = bi; __syncthreads();
    for (int st = 512; st > 0; st >>= 1) {
        if (threadIdx.x < st && rv[threadIdx.x + st] > rv[threadIdx.x]) {
            rv[threadIdx.x] = rv[threadIdx.x + st];
            ri[threadIdx.x] = ri[threadIdx.x + st];
        }
        __syncthreads();
    }
    if (threadIdx.x == 0) {
        int Bi = ri[0];
        cur[b] = Bi;
        mask[b * NI_ + Bi] = 1;
        outF[b * Q_ + k] = (float)Bi;
    }
}

// ---------------------------------------------------------------------------
extern "C" void kernel_launch(void* const* d_in, const int* in_sizes, int n_in,
                              void* d_out, int out_size, void* d_ws, size_t ws_size,
                              hipStream_t stream) {
    const float* imgF     = (const float*)d_in[0];
    const float* txtF     = (const float*)d_in[1];
    const float* tp_gamma = (const float*)d_in[2];
    const float* tp_beta  = (const float*)d_in[3];
    const float* tp_w     = (const float*)d_in[4];
    const float* tp_b     = (const float*)d_in[5];
    const float* ip_gamma = (const float*)d_in[6];
    const float* ip_beta  = (const float*)d_in[7];
    const float* ip_w     = (const float*)d_in[8];
    const float* ip_b     = (const float*)d_in[9];
    const float* tw_w1    = (const float*)d_in[10];
    const float* tw_b1    = (const float*)d_in[11];
    const float* tw_w2    = (const float*)d_in[12];
    const float* tw_b2    = (const float*)d_in[13];
    const float* temp     = (const float*)d_in[14];
    const float* lam      = (const float*)d_in[15];
    float* out = (float*)d_out;

    char* ws = (char*)d_ws;
    size_t off = 0;
    auto alloc = [&](size_t bytes) -> char* {
        char* p = ws + off;
        off = (off + bytes + 255) & ~(size_t)255;
        return p;
    };
    __bf16* ln_text  = (__bf16*)alloc((size_t)B_ * NT_ * H_ * 2);
    __bf16* raw_text = (__bf16*)alloc((size_t)B_ * NT_ * H_ * 2);
    __bf16* ln_img   = (__bf16*)alloc((size_t)B_ * NI_ * H_ * 2);
    __bf16* packTP   = (__bf16*)alloc((size_t)H_ * D_ * 2);
    __bf16* packTW   = (__bf16*)alloc((size_t)H_ * D_ * 2);
    __bf16* packIP   = (__bf16*)alloc((size_t)H_ * D_ * 2);
    float*  text_c   = (float*) alloc((size_t)B_ * NT_ * D_ * 4);
    float*  twh      = (float*) alloc((size_t)B_ * NT_ * D_ * 4);
    float*  tw_raw   = (float*) alloc((size_t)B_ * NT_ * 4);
    float*  wtf      = (float*) alloc((size_t)B_ * NT_ * D_ * 4);
    __bf16* wtb      = (__bf16*)alloc((size_t)B_ * NT_ * D_ * 2);
    __bf16* img_c    = (__bf16*)alloc((size_t)B_ * NI_ * D_ * 2);
    float*  logits   = (float*) alloc((size_t)B_ * NI_ * NT_ * 4);   // 67 MB (L2-resident)
    __bf16* Gb       = (__bf16*)alloc((size_t)B_ * D_ * D_ * 2);
    float*  P        = (float*) alloc((size_t)B_ * NI_ * D_ * 4);    // 8 MB
    float*  colmax   = (float*) alloc((size_t)B_ * NT_ * 4);
    float*  colsum   = (float*) alloc((size_t)B_ * NT_ * 4);
    float*  x2       = (float*) alloc((size_t)B_ * NI_ * 4);
    float*  logs     = (float*) alloc((size_t)B_ * NI_ * 4);
    float*  scores   = (float*) alloc((size_t)B_ * NI_ * 4);
    float*  sumd     = (float*) alloc((size_t)B_ * NI_ * 4);
    int*    mask     = (int*)   alloc((size_t)B_ * NI_ * 4);
    int*    cur      = (int*)   alloc((size_t)B_ * 4);

    // Phase 1: LayerNorms + weight packing (bf16 fragment-major)
    k_layernorm<<<B_ * NT_, 256, 0, stream>>>(txtF, tp_gamma, tp_beta, ln_text, raw_text);
    k_layernorm<<<B_ * NI_, 256, 0, stream>>>(imgF, ip_gamma, ip_beta, ln_img, nullptr);
    k_pack_w<<<192, 256, 0, stream>>>(tp_w,  packTP);
    k_pack_w<<<192, 256, 0, stream>>>(tw_w1, packTW);
    k_pack_w<<<192, 256, 0, stream>>>(ip_w,  packIP);

    // Phase 2: WMMA projections (bf16 x bf16 -> f32 accumulate)
    k_gemm768x64<<<(B_ * NT_) / 128, 256, 0, stream>>>(ln_text,  packTP, tp_b,  txtF, text_c, nullptr, 0);
    k_gemm768x64<<<(B_ * NT_) / 128, 256, 0, stream>>>(raw_text, packTW, tw_b1, nullptr, twh,  nullptr, 1);
    k_gemm768x64<<<(B_ * NI_) / 128, 256, 0, stream>>>(ln_img,   packIP, ip_b,  imgF, nullptr, img_c,  2);

    // Phase 3: token-weight tail + softmax weighting of text_c
    k_tw2<<<(B_ * NT_ + 255) / 256, 256, 0, stream>>>(twh, tw_w2, tw_b2, tw_raw);
    k_tw_softmax<<<B_, 256, 0, stream>>>(tw_raw, text_c, wtf, wtb);

    // Phase 4: logits GEMM (WMMA) + softmax stats + scores
    k_logits<<<2048, 256, 0, stream>>>(img_c, wtb, temp, logits);
    k_colstats<<<B_ * NT_, 256, 0, stream>>>(logits, colmax, colsum);
    k_scores<<<(B_ * NI_) / 8, 256, 0, stream>>>(logits, colmax, colsum,
                                                 x2, logs, scores, sumd, mask);

    // Phase 5: Gram factorization for O(NI*D) selection steps
    k_gram<<<B_, 256, 0, stream>>>(wtf, temp, Gb);
    k_gemmP<<<256, 256, 0, stream>>>(img_c, Gb, P);

    // Phase 6: greedy diverse selection (incremental distance sums)
    k_first<<<B_, 256, 0, stream>>>(scores, out, cur, mask);
    for (int k = 1; k < Q_; k++)
        k_step<<<B_, 1024, 0, stream>>>(P, img_c, x2, logs, sumd, mask, cur, lam, out, k);
}
